// GraphSAGEModel_2783138808356
// MI455X (gfx1250) — matrix operand order repaired and checked by reading, not compiled, not run on hardware
//
#include <hip/hip_runtime.h>

typedef __bf16 bf16;
typedef __attribute__((ext_vector_type(16))) __bf16 v16bf;
typedef __attribute__((ext_vector_type(8)))  __bf16 v8bf;
typedef __attribute__((ext_vector_type(4)))  __bf16 v4bf;
typedef __attribute__((ext_vector_type(8)))  float  v8f;
typedef __attribute__((ext_vector_type(4)))  unsigned v4u;
typedef __attribute__((ext_vector_type(8)))  unsigned v8u;

#define CIN   128
#define KT    256          // fused K = 2*CIN  (mean_nbr | self)
#define BLK_M 64           // nodes per block

__device__ __forceinline__ void atomAddF(float* p, float v) {
  __hip_atomic_fetch_add(p, v, __ATOMIC_RELAXED, __HIP_MEMORY_SCOPE_AGENT);
}

__global__ __launch_bounds__(256) void k_zero(float* __restrict__ p, int n) {
  int i = blockIdx.x * 256 + threadIdx.x;
  if (i < n) p[i] = 0.0f;
}

__global__ __launch_bounds__(256) void k_cvt_bf16(const float* __restrict__ in,
                                                  bf16* __restrict__ out, int n) {
  int i = blockIdx.x * 256 + threadIdx.x;
  if (i < n) out[i] = (bf16)in[i];
}

// Wcat[c][k] = k < CIN ? Wn[c][k] : Ws[c][k-CIN], converted to bf16
__global__ __launch_bounds__(256) void k_wcat(const float* __restrict__ wn,
                                              const float* __restrict__ ws,
                                              bf16* __restrict__ wcat, int cout) {
  int i = blockIdx.x * 256 + threadIdx.x;
  if (i >= cout * KT) return;
  int c = i >> 8, k = i & 255;
  float v = (k < CIN) ? wn[c * CIN + k] : ws[c * CIN + (k - CIN)];
  wcat[i] = (bf16)v;
}

__global__ __launch_bounds__(256) void k_degree(const int* __restrict__ ei,
                                                float* __restrict__ deg, int nE) {
  int e = blockIdx.x * 256 + threadIdx.x;
  if (e < nE) atomAddF(&deg[ei[nE + e]], 1.0f);
}

// one wave per edge; each lane handles 4 channels (32*4 = 128)
__global__ __launch_bounds__(256) void k_scatter(const bf16* __restrict__ xin,
                                                 const int* __restrict__ ei,
                                                 float* __restrict__ agg, int nE) {
  long long idx = (long long)blockIdx.x * 256 + threadIdx.x;
  int e = (int)(idx >> 5);
  if (e >= nE) return;
  int c4 = ((int)idx & 31) * 4;
  int src = ei[e];
  int dst = ei[nE + e];
  v4bf v = *(const v4bf*)(xin + (size_t)src * CIN + c4);
  float* ap = agg + (size_t)dst * CIN + c4;
  atomAddF(ap + 0, (float)v[0]);
  atomAddF(ap + 1, (float)v[1]);
  atomAddF(ap + 2, (float)v[2]);
  atomAddF(ap + 3, (float)v[3]);
}

// out[n,c] = relu?( [mean_nbr | x] @ Wcat^T + bias )
template<int COUT, bool RELU, bool OUTF32>
__global__ __launch_bounds__(256)
void k_sage_gemm(const float* __restrict__ agg, const float* __restrict__ deg,
                 const bf16* __restrict__ xin, const bf16* __restrict__ wcat,
                 const float* __restrict__ bias,
                 bf16* __restrict__ outb, float* __restrict__ outf, int nNodes)
{
  constexpr int NT_C = COUT / 16;                      // 8 or 4
  constexpr int MTILES = BLK_M / 16;                   // 4
  constexpr int WAVES_PER_CT = 8 / NT_C;               // 1 or 2
  constexpr int MT_PER_WAVE = MTILES / WAVES_PER_CT;   // 4 or 2
  constexpr int LDK = KT + 8;                          // +16B pad: kills bank conflicts

  __shared__ __align__(16) bf16 As[BLK_M][LDK];        // 64 x 528B = 33 KB

  const int tid = threadIdx.x;
  const int nodeBase = blockIdx.x * BLK_M;
  const bool fullBlk = (nodeBase + BLK_M) <= nNodes;

  // ---- TDM: async copy self part (x rows) into cols [CIN,2CIN) of A panel ----
  // 2D tile: tile_dim0 = 128 elems (256B rows), tile_dim1 = 64 rows.
  // LDS pad: after every 64 DWORDs (256B) insert 68 DWORDs (272B) -> 528B row
  // stride = LDK*2, with the self part landing at row*528 + 256.
  // tensor_dim1 = remaining rows, so the tail block's OOB rows read as zero.
  if (tid < 32) {
    unsigned ldsBase = (unsigned)(unsigned long long)(uintptr_t)&As[0][0];
    unsigned long long ga =
        (unsigned long long)(uintptr_t)(xin + (size_t)nodeBase * CIN);
    unsigned rows = (unsigned)(nNodes - nodeBase);
    v4u g0;
    g0[0] = 1u;                                        // count=1, user mode
    g0[1] = ldsBase + (unsigned)(CIN * 2);             // first row at +256B
    g0[2] = (unsigned)(ga & 0xFFFFFFFFu);              // global_addr[31:0]
    g0[3] = (unsigned)((ga >> 32) & 0x01FFFFFFu) | (2u << 30);  // addr[56:32] | type=2
    v8u g1;
    g1[0] = (1u << 16)                                 // data_size = 2B
          | (1u << 20)                                 // pad_enable
          | (5u << 22)                                 // pad_interval: 64 DWORDs
          | (67u << 25);                               // pad_amount: 68 DWORDs
    g1[1] = ((unsigned)CIN & 0xFFFFu) << 16;           // tensor_dim0[15:0] = 128
    g1[2] = (rows & 0xFFFFu) << 16;                    // dim0 hi=0 | tensor_dim1 lo
    g1[3] = ((rows >> 16) & 0xFFFFu)                   // tensor_dim1 hi
          | ((unsigned)CIN << 16);                     // tile_dim0 = 128
    g1[4] = (unsigned)BLK_M;                           // tile_dim1=64, tile_dim2=0
    g1[5] = (unsigned)CIN;                             // tensor_dim0_stride = 128
    g1[6] = 0u;
    g1[7] = 0u;
    asm volatile("tensor_load_to_lds %0, %1" :: "s"(g0), "s"(g1) : "memory");
  }

  // ---- stage mean-neighbor part: cols [0,CIN) = agg/deg (overlaps TDM) ----
  for (int i = tid; i < BLK_M * (CIN / 4); i += 256) {
    int r = i / (CIN / 4);
    int c4 = (i % (CIN / 4)) * 4;
    int node = nodeBase + r;
    float4 v = make_float4(0.f, 0.f, 0.f, 0.f);
    float rd = 0.f;
    if (node < nNodes) {
      v = *(const float4*)(agg + (size_t)node * CIN + c4);
      rd = 1.0f / fmaxf(deg[node], 1.0f);
    }
    As[r][c4 + 0] = (bf16)(v.x * rd);
    As[r][c4 + 1] = (bf16)(v.y * rd);
    As[r][c4 + 2] = (bf16)(v.z * rd);
    As[r][c4 + 3] = (bf16)(v.w * rd);
  }

  if (tid < 32) __builtin_amdgcn_s_wait_tensorcnt(0);  // wave 0: TDM done
  __syncthreads();                                     // publish A panel

  const int wave = tid >> 5;
  const int lane = tid & 31;
  const int l16  = lane & 15;
  const int sel  = lane >> 4;
  const int ctile  = wave % NT_C;
  const int mstart = (wave / NT_C) * MT_PER_WAVE;

  // ---- B fragments (Wcat^T), kept in registers across node tiles ----
  // B 32x16 bf16: lane holds col N=l16, K rows [kbase+sel*16, +16) contiguous
  v16bf bfrag[KT / 32];
  {
    const bf16* wrow = wcat + (size_t)(ctile * 16 + l16) * KT + sel * 16;
#pragma unroll
    for (int ks = 0; ks < KT / 32; ks++) {
      v8bf lo = *(const v8bf*)(wrow + ks * 32);
      v8bf hi = *(const v8bf*)(wrow + ks * 32 + 8);
#pragma unroll
      for (int e = 0; e < 8; e++) { bfrag[ks][e] = lo[e]; bfrag[ks][8 + e] = hi[e]; }
    }
  }

  const int col = ctile * 16 + l16;
  const float bv = bias[col];

  for (int mi = 0; mi < MT_PER_WAVE; mi++) {
    const int mtile = mstart + mi;
    v8f acc;
#pragma unroll
    for (int r = 0; r < 8; r++) acc[r] = 0.f;
#pragma unroll
    for (int ks = 0; ks < KT / 32; ks++) {
      // A 16x32 bf16: lane row M=l16; elems 0..15 = K[kb+sel*16..+16) contiguous
      const bf16* arow = &As[mtile * 16 + l16][ks * 32 + sel * 16];
      v8bf lo = *(const v8bf*)(arow);
      v8bf hi = *(const v8bf*)(arow + 8);
      v16bf afrag;
#pragma unroll
      for (int e = 0; e < 8; e++) { afrag[e] = lo[e]; afrag[8 + e] = hi[e]; }
      acc = __builtin_amdgcn_wmma_f32_16x16x32_bf16(
          false, afrag, false, bfrag[ks], (short)0, acc, false, false);
    }
    // C layout: elem r -> (m = r + sel*8, n = l16)
    const size_t rowBase = (size_t)(nodeBase + mtile * 16 + sel * 8);
    if (fullBlk) {
#pragma unroll
      for (int r = 0; r < 8; r++) {
        float v = acc[r] + bv;
        if (RELU) v = fmaxf(v, 0.f);
        if (OUTF32) outf[(rowBase + r) * COUT + col] = v;
        else        outb[(rowBase + r) * COUT + col] = (bf16)v;
      }
    } else {
#pragma unroll
      for (int r = 0; r < 8; r++) {
        if ((int)(rowBase + r) < nNodes) {
          float v = acc[r] + bv;
          if (RELU) v = fmaxf(v, 0.f);
          if (OUTF32) outf[(rowBase + r) * COUT + col] = v;
          else        outb[(rowBase + r) * COUT + col] = (bf16)v;
        }
      }
    }
  }
}

// one wave per node; lanes cover 64 output channels (2 each)
__global__ __launch_bounds__(256) void k_pool_sum(const float* __restrict__ h,
                                                  const int* __restrict__ batch,
                                                  float* __restrict__ sums,
                                                  float* __restrict__ cnt, int nNodes) {
  long long idx = (long long)blockIdx.x * 256 + threadIdx.x;
  int n = (int)(idx >> 5);
  if (n >= nNodes) return;
  int c2 = ((int)idx & 31) * 2;
  int g = batch[n];
  const float* hp = h + (size_t)n * 64 + c2;
  atomAddF(&sums[g * 64 + c2 + 0], hp[0]);
  atomAddF(&sums[g * 64 + c2 + 1], hp[1]);
  if (c2 == 0) atomAddF(&cnt[g], 1.0f);
}

__global__ __launch_bounds__(256) void k_pool_div(const float* __restrict__ sums,
                                                  const float* __restrict__ cnt,
                                                  float* __restrict__ out, int nG) {
  int i = blockIdx.x * 256 + threadIdx.x;
  if (i < nG * 64) out[i] = sums[i] / fmaxf(cnt[i >> 6], 1.0f);
}

extern "C" void kernel_launch(void* const* d_in, const int* in_sizes, int n_in,
                              void* d_out, int out_size, void* d_ws, size_t ws_size,
                              hipStream_t stream) {
  (void)n_in; (void)ws_size;
  const float* x    = (const float*)d_in[0];
  const int*   ei   = (const int*)d_in[1];
  const int*   batch= (const int*)d_in[2];
  const float* wn0  = (const float*)d_in[3];
  const float* ws0  = (const float*)d_in[4];
  const float* b0   = (const float*)d_in[5];
  const float* wn1  = (const float*)d_in[6];
  const float* ws1  = (const float*)d_in[7];
  const float* b1   = (const float*)d_in[8];
  const float* wn2  = (const float*)d_in[9];
  const float* ws2  = (const float*)d_in[10];
  const float* b2   = (const float*)d_in[11];
  float* out = (float*)d_out;

  const int N = in_sizes[0] / CIN;   // 100000
  const int E = in_sizes[1] / 2;     // 1600000
  const int G = out_size / 64;       // 512

  char* p = (char*)d_ws;
  auto take = [&](size_t bytes) { char* q = p; p += (bytes + 255) & ~(size_t)255; return q; };
  bf16*  xb   = (bf16*) take((size_t)N * CIN * 2);
  bf16*  h0b  = (bf16*) take((size_t)N * CIN * 2);
  float* agg  = (float*)take((size_t)N * CIN * 4);
  float* h2   = (float*)take((size_t)N * 64 * 4);
  float* deg  = (float*)take((size_t)N * 4);
  bf16*  wc0  = (bf16*) take((size_t)128 * KT * 2);
  bf16*  wc1  = (bf16*) take((size_t)128 * KT * 2);
  bf16*  wc2  = (bf16*) take((size_t)64 * KT * 2);
  float* sums = (float*)take((size_t)G * 64 * 4);
  float* cnt  = (float*)take((size_t)G * 4);

  auto cdiv = [](long long a, long long b) { return (int)((a + b - 1) / b); };
  const int gemmBlocks = cdiv(N, BLK_M);

  // prep: bf16 activations + concatenated bf16 weights + degrees
  k_cvt_bf16<<<cdiv((long long)N * CIN, 256), 256, 0, stream>>>(x, xb, N * CIN);
  k_wcat<<<cdiv(128 * KT, 256), 256, 0, stream>>>(wn0, ws0, wc0, 128);
  k_wcat<<<cdiv(128 * KT, 256), 256, 0, stream>>>(wn1, ws1, wc1, 128);
  k_wcat<<<cdiv(64 * KT, 256), 256, 0, stream>>>(wn2, ws2, wc2, 64);
  k_zero<<<cdiv(N, 256), 256, 0, stream>>>(deg, N);
  k_degree<<<cdiv(E, 256), 256, 0, stream>>>(ei, deg, E);

  // layer 0: x -> h0
  k_zero<<<cdiv((long long)N * CIN, 256), 256, 0, stream>>>(agg, N * CIN);
  k_scatter<<<cdiv((long long)E * 32, 256), 256, 0, stream>>>(xb, ei, agg, E);
  k_sage_gemm<128, true, false><<<gemmBlocks, 256, 0, stream>>>(agg, deg, xb, wc0, b0, h0b, nullptr, N);

  // layer 1: h0 -> xb (reuse)
  k_zero<<<cdiv((long long)N * CIN, 256), 256, 0, stream>>>(agg, N * CIN);
  k_scatter<<<cdiv((long long)E * 32, 256), 256, 0, stream>>>(h0b, ei, agg, E);
  k_sage_gemm<128, true, false><<<gemmBlocks, 256, 0, stream>>>(agg, deg, h0b, wc1, b1, xb, nullptr, N);

  // layer 2: xb -> h2 (f32, no relu)
  k_zero<<<cdiv((long long)N * CIN, 256), 256, 0, stream>>>(agg, N * CIN);
  k_scatter<<<cdiv((long long)E * 32, 256), 256, 0, stream>>>(xb, ei, agg, E);
  k_sage_gemm<64, false, true><<<gemmBlocks, 256, 0, stream>>>(agg, deg, xb, wc2, b2, nullptr, h2, N);

  // global mean pool
  k_zero<<<cdiv(G * 64, 256), 256, 0, stream>>>(sums, G * 64);
  k_zero<<<cdiv(G, 256), 256, 0, stream>>>(cnt, G);
  k_pool_sum<<<cdiv((long long)N * 32, 256), 256, 0, stream>>>(h2, batch, sums, cnt, N);
  k_pool_div<<<cdiv(G * 64, 256), 256, 0, stream>>>(sums, cnt, out, G);
}
